// UnetLayer_50663434224179
// MI455X (gfx1250) — matrix operand order repaired and checked by reading, not compiled, not run on hardware
//
#include <hip/hip_runtime.h>
#include <math.h>

// ---------------- problem constants ----------------
constexpr int NB   = 8;     // batch
constexpr int NC   = 256;   // channels / tokens-per-batch for MoE
constexpr int NL   = 512;   // length / d for MoE
constexpr int NG   = 32;    // groupnorm groups
constexpr int NH   = 8;     // heads
constexpr int NDH  = 32;    // head dim
constexpr int NE   = 1024;  // experts
constexpr int NHID = 32;    // expert hidden
constexpr int NCAP = 4;     // expert capacity

// ws layout (float offsets)
constexpr size_t OFF_XRES = 0;
constexpr size_t OFF_ACT  = 1u * 1048576;
constexpr size_t OFF_T1   = 2u * 1048576;
constexpr size_t OFF_CUR  = 3u * 1048576;
constexpr size_t OFF_BIG  = 4u * 1048576;   // 3M floats (qkv / logits / y_full)
constexpr size_t OFF_ROUTE= 7340032;        // routing + loss scratch

typedef __attribute__((ext_vector_type(16))) _Float16 v16h;
typedef __attribute__((ext_vector_type(8)))  float    v8f;

__device__ __forceinline__ float geluf(float x) {
    return 0.5f * x * (1.0f + erff(x * 0.70710678118654752f));
}

__device__ __forceinline__ v8f wmma32(v16h a, v16h b, v8f c) {
    return __builtin_amdgcn_wmma_f32_16x16x32_f16(false, a, false, b, (short)0, c, false, false);
}

// pack 4 float4 runs (contiguous K) into an A/B fragment half-pair
__device__ __forceinline__ void frag_from_f4(v16h& f, float4 x0, float4 x1, float4 x2, float4 x3) {
    f[0]=(_Float16)x0.x; f[1]=(_Float16)x0.y; f[2]=(_Float16)x0.z; f[3]=(_Float16)x0.w;
    f[4]=(_Float16)x1.x; f[5]=(_Float16)x1.y; f[6]=(_Float16)x1.z; f[7]=(_Float16)x1.w;
    f[8]=(_Float16)x2.x; f[9]=(_Float16)x2.y; f[10]=(_Float16)x2.z; f[11]=(_Float16)x2.w;
    f[12]=(_Float16)x3.x; f[13]=(_Float16)x3.y; f[14]=(_Float16)x3.z; f[15]=(_Float16)x3.w;
}

// =====================================================================
// Generic GEMM: out[M,N] = A * W[K,N] (+bias), f16 WMMA, f32 accumulate.
// AMODE 0: A row-major [M,K] (vectorized b128 fragment loads).
// AMODE 1: A is (B,C,L): row=(b,l), k=c (strided scalar loads).
// OMODE 0: out row-major [M,N]. OMODE 1: out is (B,N,L).
// One wave per 32x32 output tile (4 WMMAs / k-step), 4 waves/block.
// =====================================================================
template<int AMODE, int OMODE>
__global__ __launch_bounds__(128)
void k_gemm_wmma(const float* __restrict__ A, const float* __restrict__ W,
                 const float* __restrict__ bias, float* __restrict__ out,
                 int M, int N, int K) {
    int wave = blockIdx.x * (blockDim.x >> 5) + (threadIdx.x >> 5);
    int lane = threadIdx.x & 31;
    int lr = lane & 15, lh = lane >> 4;
    int ntn = N >> 5;
    int tm = wave / ntn, tn = wave % ntn;
    int row0 = tm * 32 + lr, row1 = row0 + 16;
    int col0 = tn * 32 + lr, col1 = col0 + 16;
    int b0_ = row0 / NL, l0_ = row0 % NL;
    int b1_ = row1 / NL, l1_ = row1 % NL;

    v8f acc00, acc01, acc10, acc11;
    float bv0 = bias ? bias[col0] : 0.0f;
    float bv1 = bias ? bias[col1] : 0.0f;
#pragma unroll
    for (int r = 0; r < 8; ++r) { acc00[r]=bv0; acc01[r]=bv1; acc10[r]=bv0; acc11[r]=bv1; }

    for (int k0 = 0; k0 < K; k0 += 32) {
        v16h a0, a1, b0, b1;
        if (AMODE == 0) {
            const float* ap0 = A + (size_t)row0 * K + k0;
            const float* ap1 = A + (size_t)row1 * K + k0;
            frag_from_f4(a0, *(const float4*)(ap0 + 8*lh),      *(const float4*)(ap0 + 8*lh + 4),
                             *(const float4*)(ap0 + 16 + 8*lh), *(const float4*)(ap0 + 16 + 8*lh + 4));
            frag_from_f4(a1, *(const float4*)(ap1 + 8*lh),      *(const float4*)(ap1 + 8*lh + 4),
                             *(const float4*)(ap1 + 16 + 8*lh), *(const float4*)(ap1 + 16 + 8*lh + 4));
        } else {
#pragma unroll
            for (int e = 0; e < 16; ++e) {
                int ka = k0 + ((e < 8) ? (8 * lh + e) : (8 + 8 * lh + e));
                a0[e] = (_Float16)A[(size_t)b0_ * K * NL + (size_t)ka * NL + l0_];
                a1[e] = (_Float16)A[(size_t)b1_ * K * NL + (size_t)ka * NL + l1_];
            }
        }
#pragma unroll
        for (int e = 0; e < 16; ++e) {
            int kb = k0 + ((e < 8) ? (8 * lh + e) : (8 + 8 * lh + e));
            b0[e] = (_Float16)W[(size_t)kb * N + col0];
            b1[e] = (_Float16)W[(size_t)kb * N + col1];
        }
        if (k0 + 32 < K)   // speculative prefetch of next weight rows (global_prefetch_b8)
            __builtin_prefetch(W + (size_t)(k0 + 32 + lane) * N + tn * 32, 0, 0);

        acc00 = wmma32(a0, b0, acc00);
        acc01 = wmma32(a0, b1, acc01);
        acc10 = wmma32(a1, b0, acc10);
        acc11 = wmma32(a1, b1, acc11);
    }

#pragma unroll
    for (int r = 0; r < 8; ++r) {
        int mA = tm * 32 + 8 * lh + r, mB = mA + 16;
        int nA = tn * 32 + lr,         nB = nA + 16;
        if (OMODE == 0) {
            out[(size_t)mA * N + nA] = acc00[r];
            out[(size_t)mA * N + nB] = acc01[r];
            out[(size_t)mB * N + nA] = acc10[r];
            out[(size_t)mB * N + nB] = acc11[r];
        } else {
            int bA = mA / NL, lA = mA % NL, bB = mB / NL, lB = mB % NL;
            out[(size_t)bA * N * NL + (size_t)nA * NL + lA] = acc00[r];
            out[(size_t)bA * N * NL + (size_t)nB * NL + lA] = acc01[r];
            out[(size_t)bB * N * NL + (size_t)nA * NL + lB] = acc10[r];
            out[(size_t)bB * N * NL + (size_t)nB * NL + lB] = acc11[r];
        }
    }
}

// =====================================================================
// conv1d k=3 pad=1 as implicit GEMM, K = 3*CIN, one wave per 32x32 tile.
// Boundary handling is branchless: tap index t = k0>>8 is uniform per
// k-step, so the pad check is one clamp+select per lane per step.
// act: (B,CIN,L), w: (Cout,CIN,3), out: (B,Cout,L) (+res same layout)
// =====================================================================
template<int CIN, bool HASRES>
__global__ __launch_bounds__(128)
void k_conv3_wmma(const float* __restrict__ act, const float* __restrict__ w,
                  const float* __restrict__ bias, const float* __restrict__ res,
                  float* __restrict__ out, int Cout) {
    int wave = blockIdx.x * (blockDim.x >> 5) + (threadIdx.x >> 5);
    int lane = threadIdx.x & 31;
    int lr = lane & 15, lh = lane >> 4;
    int ntn = Cout >> 5;
    int tm = wave / ntn, tn = wave % ntn;
    int row0 = tm * 32 + lr, row1 = row0 + 16;
    int b0_ = row0 / NL, l0_ = row0 % NL;
    int b1_ = row1 / NL, l1_ = row1 % NL;
    int col0 = tn * 32 + lr, col1 = col0 + 16;
    const int K = 3 * CIN;

    v8f acc00, acc01, acc10, acc11;
    float bv0 = bias[col0], bv1 = bias[col1];
#pragma unroll
    for (int r = 0; r < 8; ++r) { acc00[r]=bv0; acc01[r]=bv1; acc10[r]=bv0; acc11[r]=bv1; }

    for (int k0 = 0; k0 < K; k0 += 32) {
        int ta = k0 >> 8;              // uniform within this k-step (32 | 256)
        int ibase = k0 & (CIN - 1);
        // branchless pad: clamp l, mask value
        int la0 = l0_ + ta - 1, la1 = l1_ + ta - 1;
        int lc0 = la0 < 0 ? 0 : (la0 > NL - 1 ? NL - 1 : la0);
        int lc1 = la1 < 0 ? 0 : (la1 > NL - 1 ? NL - 1 : la1);
        float m0 = (la0 == lc0) ? 1.0f : 0.0f;
        float m1 = (la1 == lc1) ? 1.0f : 0.0f;
        const float* ab0 = act + (size_t)b0_ * CIN * NL + lc0;
        const float* ab1 = act + (size_t)b1_ * CIN * NL + lc1;

        v16h a0, a1, b0, b1;
#pragma unroll
        for (int e = 0; e < 16; ++e) {
            int ioff = ibase + ((e < 8) ? (8 * lh + e) : (8 + 8 * lh + e));
            a0[e] = (_Float16)(ab0[(size_t)ioff * NL] * m0);
            a1[e] = (_Float16)(ab1[(size_t)ioff * NL] * m1);
        }
#pragma unroll
        for (int e = 0; e < 16; ++e) {
            int ib = ibase + ((e < 8) ? (8 * lh + e) : (8 + 8 * lh + e));
            b0[e] = (_Float16)w[(size_t)col0 * (CIN * 3) + (size_t)ib * 3 + ta];
            b1[e] = (_Float16)w[(size_t)col1 * (CIN * 3) + (size_t)ib * 3 + ta];
        }
        if (k0 + 32 < K)
            __builtin_prefetch(w + (size_t)col0 * (CIN * 3) + (size_t)(ibase + 32) * 3, 0, 0);

        acc00 = wmma32(a0, b0, acc00);
        acc01 = wmma32(a0, b1, acc01);
        acc10 = wmma32(a1, b0, acc10);
        acc11 = wmma32(a1, b1, acc11);
    }

#pragma unroll
    for (int r = 0; r < 8; ++r) {
        int mA = tm * 32 + 8 * lh + r, mB = mA + 16;
        int nA = tn * 32 + lr,         nB = nA + 16;
        int bA = mA / NL, lA = mA % NL, bB = mB / NL, lB = mB % NL;
        size_t iAA = (size_t)bA * Cout * NL + (size_t)nA * NL + lA;
        size_t iAB = (size_t)bA * Cout * NL + (size_t)nB * NL + lA;
        size_t iBA = (size_t)bB * Cout * NL + (size_t)nA * NL + lB;
        size_t iBB = (size_t)bB * Cout * NL + (size_t)nB * NL + lB;
        float vAA = acc00[r], vAB = acc01[r], vBA = acc10[r], vBB = acc11[r];
        if (HASRES) { vAA += res[iAA]; vAB += res[iAB]; vBA += res[iBA]; vBB += res[iBB]; }
        out[iAA] = vAA; out[iAB] = vAB; out[iBA] = vBA; out[iBB] = vBB;
    }
}

// =====================================================================
// (optional x+emb) -> GroupNorm -> affine -> GELU.  One block per (b,g).
// =====================================================================
template<bool HASEMB>
__global__ void k_gn_gelu(const float* __restrict__ x, const float* __restrict__ emb,
                          const float* __restrict__ scale, const float* __restrict__ bias,
                          float* __restrict__ xres, float* __restrict__ act) {
    int b = blockIdx.x / NG, g = blockIdx.x % NG;
    const int CPG = NC / NG;
    int tid = threadIdx.x;
    __shared__ float red[256];
    float v[16];
    float s = 0.f, s2 = 0.f;
#pragma unroll
    for (int j = 0; j < 16; ++j) {
        int i = tid + 256 * j;
        int c = g * CPG + i / NL, l = i % NL;
        size_t idx = (size_t)b * NC * NL + (size_t)c * NL + l;
        float val = x[idx];
        if (HASEMB) { val += emb[idx]; xres[idx] = val; }
        v[j] = val; s += val; s2 += val * val;
    }
    red[tid] = s; __syncthreads();
    for (int st = 128; st > 0; st >>= 1) { if (tid < st) red[tid] += red[tid + st]; __syncthreads(); }
    float mean = red[0] * (1.0f / 4096.0f); __syncthreads();
    red[tid] = s2; __syncthreads();
    for (int st = 128; st > 0; st >>= 1) { if (tid < st) red[tid] += red[tid + st]; __syncthreads(); }
    float var = red[0] * (1.0f / 4096.0f) - mean * mean;
    float inv = rsqrtf(var + 1e-5f);
#pragma unroll
    for (int j = 0; j < 16; ++j) {
        int i = tid + 256 * j;
        int c = g * CPG + i / NL, l = i % NL;
        size_t idx = (size_t)b * NC * NL + (size_t)c * NL + l;
        float y = (v[j] - mean) * inv * scale[c] + bias[c];
        act[idx] = geluf(y);
    }
}

// =====================================================================
// Attention core: one thread per (b,h,query). dh=32, L=512, online softmax.
// qkv: (B,L,768), f = d*24 + h*3 + sel.  tmp out: (B,L,C), c = d*8 + h.
// =====================================================================
__global__ void k_attn(const float* __restrict__ qkv, float* __restrict__ tmp) {
    int idx = blockIdx.x * blockDim.x + threadIdx.x;
    int l = idx % NL;
    int h = (idx / NL) % NH;
    int b = idx / (NL * NH);
    const float* base = qkv + (size_t)b * NL * 768 + (size_t)l * 768;
    float q[NDH], o[NDH];
#pragma unroll
    for (int d = 0; d < NDH; ++d) { q[d] = base[d * 24 + h * 3 + 0]; o[d] = 0.f; }
    float m = -3.0e38f, ssum = 0.f;
    const float scl = 0.17677669529663687f;
    for (int k = 0; k < NL; ++k) {
        const float* kb = qkv + (size_t)b * NL * 768 + (size_t)k * 768;
        float dot = 0.f;
#pragma unroll
        for (int d = 0; d < NDH; ++d) dot += q[d] * kb[d * 24 + h * 3 + 1];
        dot *= scl;
        float mn = fmaxf(m, dot);
        float corr = __expf(m - mn);
        float p = __expf(dot - mn);
        ssum = ssum * corr + p;
#pragma unroll
        for (int d = 0; d < NDH; ++d) o[d] = o[d] * corr + p * kb[d * 24 + h * 3 + 2];
        m = mn;
    }
    float invs = 1.0f / ssum;
#pragma unroll
    for (int d = 0; d < NDH; ++d)
        tmp[(size_t)b * NL * NC + (size_t)l * NC + d * NH + h] = o[d] * invs;
}

// =====================================================================
// MoE top-2 gating with capacity (exact exclusive-cumsum semantics via
// running per-expert counters). One block per batch, serial over tokens.
// =====================================================================
__global__ void k_moe_gate(const float* __restrict__ logits,
                           int* __restrict__ tok_e1, int* __restrict__ tok_e2,
                           float* __restrict__ tok_g1, float* __restrict__ tok_g2,
                           float* __restrict__ lossb) {
    int b = blockIdx.x, tid = threadIdx.x;
    __shared__ float sprob[NE];
    __shared__ float sumraw[NE];
    __shared__ int   cntA[NE];
    __shared__ int   cntCap[NE];
    __shared__ int   cntB[NE];
    __shared__ float rv[256];
    __shared__ int   ri[256];
    for (int e = tid; e < NE; e += 256) { sumraw[e] = 0.f; cntA[e] = 0; cntCap[e] = 0; cntB[e] = 0; }
    __syncthreads();

    for (int t = 0; t < NC; ++t) {
        const float* row = logits + ((size_t)b * NC + t) * NE;
        float mx = -3.0e38f;
        for (int e = tid; e < NE; e += 256) mx = fmaxf(mx, row[e]);
        rv[tid] = mx; __syncthreads();
        for (int st = 128; st > 0; st >>= 1) { if (tid < st) rv[tid] = fmaxf(rv[tid], rv[tid + st]); __syncthreads(); }
        float M = rv[0]; __syncthreads();
        float sl = 0.f;
        for (int e = tid; e < NE; e += 256) { float p = __expf(row[e] - M); sprob[e] = p; sl += p; }
        rv[tid] = sl; __syncthreads();
        for (int st = 128; st > 0; st >>= 1) { if (tid < st) rv[tid] += rv[tid + st]; __syncthreads(); }
        float inv = 1.0f / rv[0]; __syncthreads();
        float bv = -1.f; int bi = NE;
        for (int e = tid; e < NE; e += 256) {
            float p = sprob[e] * inv; sprob[e] = p; sumraw[e] += p;
            if (p > bv || (p == bv && e < bi)) { bv = p; bi = e; }
        }
        __syncthreads();
        rv[tid] = bv; ri[tid] = bi; __syncthreads();
        for (int st = 128; st > 0; st >>= 1) {
            if (tid < st && (rv[tid + st] > rv[tid] || (rv[tid + st] == rv[tid] && ri[tid + st] < ri[tid]))) {
                rv[tid] = rv[tid + st]; ri[tid] = ri[tid + st];
            }
            __syncthreads();
        }
        int idx1 = ri[0]; float g1 = rv[0]; __syncthreads();
        bv = -1.f; bi = NE;
        for (int e = tid; e < NE; e += 256) {
            if (e == idx1) continue;
            float p = sprob[e];
            if (p > bv || (p == bv && e < bi)) { bv = p; bi = e; }
        }
        rv[tid] = bv; ri[tid] = bi; __syncthreads();
        for (int st = 128; st > 0; st >>= 1) {
            if (tid < st && (rv[tid + st] > rv[tid] || (rv[tid + st] == rv[tid] && ri[tid + st] < ri[tid]))) {
                rv[tid] = rv[tid + st]; ri[tid] = ri[tid + st];
            }
            __syncthreads();
        }
        int idx2 = ri[0]; float g2 = rv[0]; __syncthreads();
        if (tid == 0) {
            float denom = g1 + g2 + 1e-9f;
            float g1n = g1 / denom, g2n = g2 / denom;
            int p1 = cntA[idx1]; cntA[idx1] = p1 + 1;
            float g1e = 0.f;
            if (p1 < NCAP) { cntCap[idx1]++; g1e = g1n; }
            int bt = b * NC + t;
            tok_e1[bt] = idx1; tok_g1[bt] = g1e;
            tok_e2[bt] = idx2; tok_g2[bt] = g2n;
        }
        __syncthreads();
    }
    if (tid == 0) {
        for (int t = 0; t < NC; ++t) {
            int bt = b * NC + t;
            int e2 = tok_e2[bt];
            int p2 = cntB[e2] + cntCap[e2];
            cntB[e2]++;
            if (p2 >= NCAP) tok_g2[bt] = 0.f;
        }
    }
    __syncthreads();
    float part = 0.f;
    for (int e = tid; e < NE; e += 256) part += sumraw[e] * (float)cntA[e];
    rv[tid] = part; __syncthreads();
    for (int st = 128; st > 0; st >>= 1) { if (tid < st) rv[tid] += rv[tid + st]; __syncthreads(); }
    if (tid == 0) lossb[b] = rv[0];
}

__global__ void k_loss(const float* __restrict__ lossb, float* __restrict__ out_aux) {
    if (threadIdx.x == 0) {
        float s = 0.f;
        for (int b = 0; b < NB; ++b) s += lossb[b];
        *out_aux = s * 1.953125e-5f;   // 0.01 * E / (B*n^2)
    }
}

// =====================================================================
// Expert FFN per token (<=2 experts): 512 -> gelu(32) -> 512, weighted.
// One block (128 thr) per (b,token). Deterministic, no atomics.
// =====================================================================
__global__ void k_moe_expert(const float* __restrict__ x2,
                             const int* __restrict__ tok_e1, const int* __restrict__ tok_e2,
                             const float* __restrict__ tok_g1, const float* __restrict__ tok_g2,
                             const float* __restrict__ w1, const float* __restrict__ w2,
                             float* __restrict__ xmoe, float* __restrict__ out_x) {
    int bt = blockIdx.x;
    int b = bt / NC, tok = bt % NC;
    int tid = threadIdx.x;
    __shared__ float xs[NL];
    __shared__ float hs[NHID];
    for (int i = tid; i < NL; i += 128)
        xs[i] = x2[(size_t)b * NC * NL + (size_t)tok * NL + i];
    float acc[4] = {0.f, 0.f, 0.f, 0.f};
    int   es[2] = {tok_e1[bt], tok_e2[bt]};
    float gs[2] = {tok_g1[bt], tok_g2[bt]};
    for (int a = 0; a < 2; ++a) {
        __syncthreads();
        int e = es[a]; float g = gs[a];
        if (g == 0.f) continue;               // uniform per block
        if (tid < NHID) {
            const float* wp = w1 + (size_t)e * NL * NHID + tid;
            float d = 0.f;
            for (int k = 0; k < NL; ++k) d += xs[k] * wp[(size_t)k * NHID];
            hs[tid] = geluf(d);
        }
        __syncthreads();
#pragma unroll
        for (int j = 0; j < 4; ++j) {
            int dd = tid * 4 + j;
            float s = 0.f;
#pragma unroll
            for (int t = 0; t < NHID; ++t)
                s += hs[t] * w2[(size_t)e * NHID * NL + (size_t)t * NL + dd];
            acc[j] += g * s;
        }
    }
#pragma unroll
    for (int j = 0; j < 4; ++j) {
        int dd = tid * 4 + j;
        size_t idx = (size_t)b * NC * NL + (size_t)tok * NL + dd;
        xmoe[idx] = acc[j];
        out_x[idx] = acc[j];
    }
}

// max-pool window 3, stride 2, pad 1: (B,512,512) -> (B,512,256)
__global__ void k_maxpool(const float* __restrict__ y, float* __restrict__ out) {
    int i = blockIdx.x * blockDim.x + threadIdx.x;
    int j = i % 256;
    int o = (i / 256) % 512;
    int b = i / (256 * 512);
    const float* row = y + (size_t)b * 512 * NL + (size_t)o * NL;
    float m = -3.0e38f;
    int l0 = 2 * j - 1;
#pragma unroll
    for (int d = 0; d < 3; ++d) {
        int l = l0 + d;
        if (l >= 0 && l < NL) m = fmaxf(m, row[l]);
    }
    out[i] = m;
}

// =====================================================================
extern "C" void kernel_launch(void* const* d_in, const int* in_sizes, int n_in,
                              void* d_out, int out_size, void* d_ws, size_t ws_size,
                              hipStream_t stream) {
    const float* x      = (const float*)d_in[0];
    const float* emb    = (const float*)d_in[1];
    const float* r1g1s  = (const float*)d_in[2];
    const float* r1g1b  = (const float*)d_in[3];
    const float* r1c1w  = (const float*)d_in[4];
    const float* r1c1b  = (const float*)d_in[5];
    const float* r1g2s  = (const float*)d_in[6];
    const float* r1g2b  = (const float*)d_in[7];
    const float* r1c2w  = (const float*)d_in[8];
    const float* r1c2b  = (const float*)d_in[9];
    const float* r2g1s  = (const float*)d_in[10];
    const float* r2g1b  = (const float*)d_in[11];
    const float* r2c1w  = (const float*)d_in[12];
    const float* r2c1b  = (const float*)d_in[13];
    const float* r2g2s  = (const float*)d_in[14];
    const float* r2g2b  = (const float*)d_in[15];
    const float* r2c2w  = (const float*)d_in[16];
    const float* r2c2b  = (const float*)d_in[17];
    const float* aw1    = (const float*)d_in[18];
    const float* ab1    = (const float*)d_in[19];
    const float* aw2    = (const float*)d_in[20];
    const float* ab2    = (const float*)d_in[21];
    const float* moewg  = (const float*)d_in[22];
    const float* moew1  = (const float*)d_in[23];
    const float* moew2  = (const float*)d_in[24];
    const float* outw   = (const float*)d_in[25];
    const float* outb   = (const float*)d_in[26];

    float* ws   = (float*)d_ws;
    float* xres = ws + OFF_XRES;
    float* act  = ws + OFF_ACT;
    float* t1   = ws + OFF_T1;
    float* cur  = ws + OFF_CUR;
    float* big  = ws + OFF_BIG;
    int*   tok_e1 = (int*)(ws + OFF_ROUTE);
    int*   tok_e2 = (int*)(ws + OFF_ROUTE + 2048);
    float* tok_g1 = ws + OFF_ROUTE + 4096;
    float* tok_g2 = ws + OFF_ROUTE + 6144;
    float* lossb  = ws + OFF_ROUTE + 8192;

    float* out    = (float*)d_out;
    float* out_y  = out;
    float* out_x  = out + 1048576;
    float* out_aux= out + 2097152;

    // ---------- res block 1 ----------
    k_gn_gelu<true ><<<NB * NG, 256, 0, stream>>>(x,   emb,     r1g1s, r1g1b, xres, act);
    k_conv3_wmma<NC, false><<<256, 128, 0, stream>>>(act, r1c1w, r1c1b, nullptr, t1, NC);
    k_gn_gelu<false><<<NB * NG, 256, 0, stream>>>(t1,  nullptr, r1g2s, r1g2b, nullptr, act);
    k_conv3_wmma<NC, true ><<<256, 128, 0, stream>>>(act, r1c2w, r1c2b, xres, cur, NC);

    // ---------- attention ----------
    // qkv = x^T @ w1 + b1 : M=4096, K=256, N=768
    k_gemm_wmma<1, 0><<<768, 128, 0, stream>>>(cur, aw1, ab1, big, 4096, 768, 256);
    k_attn<<<128, 256, 0, stream>>>(big, t1);
    // proj: M=4096, K=256, N=256, out back to (B,C,L)
    k_gemm_wmma<0, 1><<<256, 128, 0, stream>>>(t1, aw2, ab2, cur, 4096, 256, 256);

    // ---------- res block 2 ----------
    k_gn_gelu<true ><<<NB * NG, 256, 0, stream>>>(cur, emb,     r2g1s, r2g1b, xres, act);
    k_conv3_wmma<NC, false><<<256, 128, 0, stream>>>(act, r2c1w, r2c1b, nullptr, t1, NC);
    k_gn_gelu<false><<<NB * NG, 256, 0, stream>>>(t1,  nullptr, r2g2s, r2g2b, nullptr, act);
    k_conv3_wmma<NC, true ><<<256, 128, 0, stream>>>(act, r2c2w, r2c2b, xres, cur, NC);

    // ---------- MoE ----------
    // gate logits: M=2048, K=512, N=1024
    k_gemm_wmma<0, 0><<<512, 128, 0, stream>>>(cur, moewg, nullptr, big, 2048, 1024, 512);
    k_moe_gate<<<NB, 256, 0, stream>>>(big, tok_e1, tok_e2, tok_g1, tok_g2, lossb);
    k_loss<<<1, 32, 0, stream>>>(lossb, out_aux);
    k_moe_expert<<<NB * NC, 128, 0, stream>>>(cur, tok_e1, tok_e2, tok_g1, tok_g2,
                                              moew1, moew2, t1, out_x);

    // ---------- output conv + maxpool ----------
    k_conv3_wmma<NC, false><<<512, 128, 0, stream>>>(t1, outw, outb, nullptr, big, 512);
    k_maxpool<<<4096, 256, 0, stream>>>(big, out_y);
}
